// RNNT_60636348285717
// MI455X (gfx1250) — matrix operand order repaired and checked by reading, not compiled, not run on hardware
//
#include <hip/hip_runtime.h>
#include <hip/hip_bf16.h>
#include <math.h>

typedef __bf16 bf16_t;
typedef __attribute__((ext_vector_type(16))) __bf16 v16bf;
typedef __attribute__((ext_vector_type(8)))  __bf16 v8bf;
typedef __attribute__((ext_vector_type(8)))  float  v8f;

// Problem constants
#define T_STEPS 128
#define U_STEPS 65          // 1 SOS + 64 label steps
#define BATCH   8
#define I_IN    240
#define KX_ENC  256         // 240 padded to multiple of 32
#define E_EMB   512
#define H_DIM   1024
#define G_DIM   4096        // 4*H
#define V_DIM   1024
#define LSTM_BLOCKS 32      // workgroups per network (32-wide hidden slices)

// ---------------------------------------------------------------------------
// WMMA helpers (CDNA5 layouts, wave32)
// ---------------------------------------------------------------------------
__device__ __forceinline__ v16bf load_a_frag(const bf16_t* p) {
    // p already points at row*ld + kk + half*8
    v8bf lo = *(const v8bf*)(p);
    v8bf hi = *(const v8bf*)(p + 16);
    v16bf a;
#pragma unroll
    for (int i = 0; i < 8; ++i) { a[i] = lo[i]; a[i + 8] = hi[i]; }
    return a;
}

__device__ __forceinline__ v8f wmma_bf16(v16bf a, v16bf b, v8f c) {
    return __builtin_amdgcn_wmma_f32_16x16x32_bf16(false, a, false, b,
                                                   (short)0, c, false, false);
}

// ---------------------------------------------------------------------------
// Init / conversion kernels
// ---------------------------------------------------------------------------
__global__ void convert_pad_kernel(const float* __restrict__ src, bf16_t* __restrict__ dst,
                                   int N, int Ksrc, int Kdst) {
    long total = (long)N * Kdst;
    for (long i = (long)blockIdx.x * blockDim.x + threadIdx.x; i < total;
         i += (long)gridDim.x * blockDim.x) {
        int n = (int)(i / Kdst);
        int k = (int)(i % Kdst);
        float v = (k < Ksrc) ? src[(long)n * Ksrc + k] : 0.0f;
        dst[i] = (bf16_t)v;
    }
}

// x [T][8][240] fp32 -> [T][16][256] bf16 (batch + K padded with zeros)
__global__ void convert_x_kernel(const float* __restrict__ src, bf16_t* __restrict__ dst) {
    long total = (long)T_STEPS * 16 * KX_ENC;
    for (long i = (long)blockIdx.x * blockDim.x + threadIdx.x; i < total;
         i += (long)gridDim.x * blockDim.x) {
        int k    = (int)(i % KX_ENC);
        long r   = i / KX_ENC;
        int brow = (int)(r % 16);
        int t    = (int)(r / 16);
        float v = 0.0f;
        if (brow < BATCH && k < I_IN)
            v = src[((long)t * BATCH + brow) * I_IN + k];
        dst[i] = (bf16_t)v;
    }
}

// decoder input [65][16][512] bf16: step 0 = SOS zeros, step u = emb[y[u-1][b]]
__global__ void embed_kernel(const float* __restrict__ emb, const int* __restrict__ y,
                             bf16_t* __restrict__ dst) {
    long total = (long)U_STEPS * 16 * E_EMB;
    for (long i = (long)blockIdx.x * blockDim.x + threadIdx.x; i < total;
         i += (long)gridDim.x * blockDim.x) {
        int k    = (int)(i % E_EMB);
        long r   = i / E_EMB;
        int brow = (int)(r % 16);
        int u    = (int)(r / 16);
        float v = 0.0f;
        if (u >= 1 && brow < BATCH) {
            int tok = y[(long)(u - 1) * BATCH + brow];
            v = emb[(long)tok * E_EMB + k];
        }
        dst[i] = (bf16_t)v;
    }
}

__global__ void zero_kernel(float4* __restrict__ p, long n4) {
    float4 z = make_float4(0.f, 0.f, 0.f, 0.f);
    for (long i = (long)blockIdx.x * blockDim.x + threadIdx.x; i < n4;
         i += (long)gridDim.x * blockDim.x)
        p[i] = z;
}

// ---------------------------------------------------------------------------
// Persistent LSTM kernel. grid = (32, 2): y=0 encoder (128 steps), y=1 decoder
// (65 steps). Each workgroup owns a 32-wide hidden slice; 256 threads = 8
// waves, wave (p,s) computes the 16-col WMMA gate tile of part p (i/f/g/o).
// Between timesteps each network synchronizes its 32 blocks with a
// generational spin barrier in device memory (replaces 128 kernel launches).
// ---------------------------------------------------------------------------
struct LstmArgs {
    const bf16_t* Wih;   // [4096][Kx]  bf16
    const bf16_t* Whh;   // [4096][1024] bf16
    const float*  bias;  // [4096] fp32
    const bf16_t* xs;    // [steps][16][Kx] bf16 (padded)
    bf16_t*       hbase; // [steps+1][16][1024]; slot 0 = h0 = 0
    float*        c;     // [8][1024] fp32 cell state (in/out)
    int           Kx;
    int           steps;
};

__global__ __launch_bounds__(256) void lstm_persistent_kernel(LstmArgs ea, LstmArgs da,
                                                              unsigned* __restrict__ bars) {
    const LstmArgs& A = (blockIdx.y == 0) ? ea : da;
    unsigned* bcount = bars + (size_t)blockIdx.y * 64;   // 256B apart per network
    unsigned* bgen   = bcount + 1;
    const unsigned nblocks = gridDim.x;

    const int tid  = threadIdx.x;
    const int wave = tid >> 5;
    const int lane = tid & 31;
    const int half = lane >> 4;   // which 16-lane half
    const int l16  = lane & 15;
    const int hb   = blockIdx.x * 32;         // hidden slice base
    const int p    = wave >> 1;               // gate part 0..3 (i,f,g,o)
    const int s    = wave & 1;                // 16-col sub-tile within slice
    const int colBase = p * H_DIM + hb + s * 16;

    __shared__ float gl[4][16][32];

    // invariant per-wave weight row pointers
    const bf16_t* WihRow = A.Wih + (size_t)(colBase + l16) * A.Kx + half * 8;
    const bf16_t* WhhRow = A.Whh + (size_t)(colBase + l16) * H_DIM + half * 8;
    // pointwise-thread constants
    const int pb   = tid >> 5;     // batch row 0..7
    const int pj   = tid & 31;     // hidden col within slice
    const int pcol = hb + pj;
    const float bi = A.bias[pcol];
    const float bf = A.bias[H_DIM + pcol];
    const float bg = A.bias[2 * H_DIM + pcol];
    const float bo = A.bias[3 * H_DIM + pcol];

    for (int t = 0; t < A.steps; ++t) {
        v8f acc = {};
        // input-to-hidden contribution
        {
            const bf16_t* X = A.xs + (size_t)t * 16 * A.Kx + (size_t)l16 * A.Kx + half * 8;
            for (int kk = 0; kk < A.Kx; kk += 32) {
                v16bf a = load_a_frag(X + kk);
                v16bf b = load_a_frag(WihRow + kk);
                acc = wmma_bf16(a, b, acc);
            }
        }
        // hidden-to-hidden contribution (h_{t-1} = slot t of hbase)
        {
            const bf16_t* Hp = A.hbase + (size_t)t * 16 * H_DIM + (size_t)l16 * H_DIM + half * 8;
#pragma unroll 4
            for (int kk = 0; kk < H_DIM; kk += 32) {
                v16bf a = load_a_frag(Hp + kk);
                v16bf b = load_a_frag(WhhRow + kk);
                acc = wmma_bf16(a, b, acc);
            }
        }

        // spill gate tile to LDS: acc[r] holds (M = r + 8*half, N = l16)
#pragma unroll
        for (int r = 0; r < 8; ++r)
            gl[p][r + 8 * half][s * 16 + l16] = acc[r];
        __syncthreads();

        // pointwise: 256 threads = 8 batches x 32 hidden cols
        {
            float gi = gl[0][pb][pj] + bi;
            float gf = gl[1][pb][pj] + bf;
            float gg = gl[2][pb][pj] + bg;
            float go = gl[3][pb][pj] + bo;
            float cprev = A.c[pb * H_DIM + pcol];
            float si = 1.0f / (1.0f + __expf(-gi));
            float sf = 1.0f / (1.0f + __expf(-gf));
            float so = 1.0f / (1.0f + __expf(-go));
            float cn = sf * cprev + si * tanhf(gg);
            float hn = so * tanhf(cn);
            A.c[pb * H_DIM + pcol] = cn;
            A.hbase[(size_t)(t + 1) * 16 * H_DIM + (size_t)pb * H_DIM + pcol] = (bf16_t)hn;
        }

        // ---- inter-block barrier for this network (generational) ----
        __threadfence();          // release this thread's h/c stores
        __syncthreads();          // also guards LDS reuse next iteration
        if (tid == 0) {
            unsigned g0 = atomicAdd(bgen, 0u);           // read gen BEFORE arriving
            unsigned arrived = atomicAdd(bcount, 1u);
            if (arrived == nblocks - 1u) {
                atomicExch(bcount, 0u);
                __threadfence();
                atomicAdd(bgen, 1u);
            } else {
                while (atomicAdd(bgen, 0u) == g0)
                    __builtin_amdgcn_s_sleep(1);
            }
            __threadfence();      // acquire other blocks' h stores
        }
        __syncthreads();
    }
}

// ---------------------------------------------------------------------------
// Joint projections: lf[t] = f_t @ Wf^T, lg[u] = g_u @ Wg^T.
// blockIdx.y = item (0..127 -> lf[t], 128..192 -> lg[u]); blockIdx.x*128 +
// wave*16 = output-column tile; M = 16 (8 real batch rows), K = 1024.
// ---------------------------------------------------------------------------
__global__ __launch_bounds__(256) void joint_kernel(const bf16_t* __restrict__ fcW, // [1024][2048] bf16
                                                    const bf16_t* __restrict__ fbase,
                                                    const bf16_t* __restrict__ gbase,
                                                    float* __restrict__ lf,
                                                    float* __restrict__ lg) {
    const int item = blockIdx.y;
    const bf16_t* Abase;
    float* outp;
    int koff;
    if (item < T_STEPS) {
        Abase = fbase + (size_t)(item + 1) * 16 * H_DIM;
        outp  = lf + (size_t)item * BATCH * V_DIM;
        koff  = 0;
    } else {
        int u = item - T_STEPS;
        Abase = gbase + (size_t)(u + 1) * 16 * H_DIM;
        outp  = lg + (size_t)u * BATCH * V_DIM;
        koff  = H_DIM;
    }
    const int tid  = threadIdx.x;
    const int wave = tid >> 5;
    const int lane = tid & 31;
    const int half = lane >> 4;
    const int l16  = lane & 15;
    const int colBase = blockIdx.x * 128 + wave * 16;

    const bf16_t* Wrow = fcW + (size_t)(colBase + l16) * (2 * H_DIM) + koff + half * 8;
    const bf16_t* Arow = Abase + (size_t)l16 * H_DIM + half * 8;

    v8f acc = {};
#pragma unroll 4
    for (int kk = 0; kk < H_DIM; kk += 32) {
        v16bf a = load_a_frag(Arow + kk);
        v16bf b = load_a_frag(Wrow + kk);
        acc = wmma_bf16(a, b, acc);
    }
    if (half == 0) {   // rows M=0..7 are the real batch
#pragma unroll
        for (int r = 0; r < 8; ++r)
            outp[(size_t)r * V_DIM + colBase + l16] = acc[r];
    }
}

// ---------------------------------------------------------------------------
// Final broadcast add: out[b][t][u][v] = lf[t][b][v] + lg[u][b][v] + fc_b[v]
// Pure HBM-streaming kernel (float4).
// ---------------------------------------------------------------------------
__global__ void assemble_kernel(const float4* __restrict__ lf4, const float4* __restrict__ lg4,
                                const float4* __restrict__ fb4, float4* __restrict__ out4,
                                long total4) {
    for (long i = (long)blockIdx.x * blockDim.x + threadIdx.x; i < total4;
         i += (long)gridDim.x * blockDim.x) {
        int  v4 = (int)(i & 255);            // 1024/4 = 256 float4 per (b,t,u)
        long q  = i >> 8;
        int  u  = (int)(q % U_STEPS); q /= U_STEPS;
        int  t  = (int)(q % T_STEPS);
        int  b  = (int)(q / T_STEPS);
        float4 a  = lf4[((long)t * BATCH + b) * 256 + v4];
        float4 g  = lg4[((long)u * BATCH + b) * 256 + v4];
        float4 bb = fb4[v4];
        out4[i] = make_float4(a.x + g.x + bb.x, a.y + g.y + bb.y,
                              a.z + g.z + bb.z, a.w + g.w + bb.w);
    }
}

// ---------------------------------------------------------------------------
// Host launcher
// ---------------------------------------------------------------------------
extern "C" void kernel_launch(void* const* d_in, const int* in_sizes, int n_in,
                              void* d_out, int out_size, void* d_ws, size_t ws_size,
                              hipStream_t stream) {
    (void)in_sizes; (void)n_in; (void)out_size; (void)ws_size;
    const float* x       = (const float*)d_in[0];
    const int*   y       = (const int*)d_in[2];
    const float* enc_Wih = (const float*)d_in[4];
    const float* enc_Whh = (const float*)d_in[5];
    const float* enc_b   = (const float*)d_in[6];
    const float* emb     = (const float*)d_in[7];
    const float* dec_Wih = (const float*)d_in[8];
    const float* dec_Whh = (const float*)d_in[9];
    const float* dec_b   = (const float*)d_in[10];
    const float* fc_W    = (const float*)d_in[11];
    const float* fc_b    = (const float*)d_in[12];

    char* ws = (char*)d_ws;
    size_t off = 0;
    auto walloc = [&](size_t bytes) -> void* {
        void* p = ws + off;
        off += (bytes + 255) & ~(size_t)255;
        return p;
    };

    bf16_t* wih_e = (bf16_t*)walloc((size_t)G_DIM * KX_ENC * 2);
    bf16_t* whh_e = (bf16_t*)walloc((size_t)G_DIM * H_DIM * 2);
    bf16_t* wih_d = (bf16_t*)walloc((size_t)G_DIM * E_EMB * 2);
    bf16_t* whh_d = (bf16_t*)walloc((size_t)G_DIM * H_DIM * 2);
    bf16_t* fcw   = (bf16_t*)walloc((size_t)V_DIM * 2 * H_DIM * 2);
    bf16_t* xb    = (bf16_t*)walloc((size_t)T_STEPS * 16 * KX_ENC * 2);
    bf16_t* din   = (bf16_t*)walloc((size_t)U_STEPS * 16 * E_EMB * 2);
    // --- zero-initialized region (state slabs + barrier words) ---
    char*     zbase = ws + off;
    bf16_t*   fbuf  = (bf16_t*)walloc((size_t)(T_STEPS + 1) * 16 * H_DIM * 2);
    bf16_t*   gbuf  = (bf16_t*)walloc((size_t)(U_STEPS + 1) * 16 * H_DIM * 2);
    float*    cenc  = (float*)walloc((size_t)BATCH * H_DIM * 4);
    float*    cdec  = (float*)walloc((size_t)BATCH * H_DIM * 4);
    unsigned* bars  = (unsigned*)walloc(128 * sizeof(unsigned));  // 2 networks x 64
    size_t    zbytes = (size_t)((ws + off) - zbase);
    // --- joint outputs ---
    float*  lf = (float*)walloc((size_t)T_STEPS * BATCH * V_DIM * 4);
    float*  lg = (float*)walloc((size_t)U_STEPS * BATCH * V_DIM * 4);

    const int thr = 256;
    auto blocks = [](long n, int t) { return (unsigned)((n + t - 1) / t); };

    convert_pad_kernel<<<blocks((long)G_DIM * KX_ENC, thr), thr, 0, stream>>>(enc_Wih, wih_e, G_DIM, I_IN, KX_ENC);
    convert_pad_kernel<<<blocks((long)G_DIM * H_DIM, thr), thr, 0, stream>>>(enc_Whh, whh_e, G_DIM, H_DIM, H_DIM);
    convert_pad_kernel<<<blocks((long)G_DIM * E_EMB, thr), thr, 0, stream>>>(dec_Wih, wih_d, G_DIM, E_EMB, E_EMB);
    convert_pad_kernel<<<blocks((long)G_DIM * H_DIM, thr), thr, 0, stream>>>(dec_Whh, whh_d, G_DIM, H_DIM, H_DIM);
    convert_pad_kernel<<<blocks((long)V_DIM * 2 * H_DIM, thr), thr, 0, stream>>>(fc_W, fcw, V_DIM, 2 * H_DIM, 2 * H_DIM);
    convert_x_kernel<<<blocks((long)T_STEPS * 16 * KX_ENC, thr), thr, 0, stream>>>(x, xb);
    embed_kernel<<<blocks((long)U_STEPS * 16 * E_EMB, thr), thr, 0, stream>>>(emb, y, din);
    zero_kernel<<<1024, thr, 0, stream>>>((float4*)zbase, (long)(zbytes / 16));

    LstmArgs ea { wih_e, whh_e, enc_b, xb,  fbuf, cenc, KX_ENC, T_STEPS };
    LstmArgs da { wih_d, whh_d, dec_b, din, gbuf, cdec, E_EMB,  U_STEPS };

    // Single persistent launch for both recurrences (64 co-resident blocks).
    lstm_persistent_kernel<<<dim3(LSTM_BLOCKS, 2), 256, 0, stream>>>(ea, da, bars);

    joint_kernel<<<dim3(8, T_STEPS + U_STEPS), 256, 0, stream>>>(fcw, fbuf, gbuf, lf, lg);

    const long total4 = (long)BATCH * T_STEPS * U_STEPS * (V_DIM / 4);
    assemble_kernel<<<8192, 256, 0, stream>>>((const float4*)lf, (const float4*)lg,
                                              (const float4*)fc_b, (float4*)d_out, total4);
}
